// MoGIN12_48266842472586
// MI455X (gfx1250) — matrix operand order repaired and checked by reading, not compiled, not run on hardware
//
#include <hip/hip_runtime.h>
#include <hip/hip_bf16.h>

#define NN 40000
#define GG 50
#define EE 640000
#define DD 128
#define DE 32
#define KK 8
#define EPSF 1e-5f

typedef __attribute__((ext_vector_type(16))) _Float16 v16h;
typedef __attribute__((ext_vector_type(8)))  float    v8f;

union HU { v16h h; unsigned u[8]; };

struct MlpP { const _Float16* W0; const _Float16* W1;
              const float* b0; const float* b1; const float* g0; const float* be0; };
struct MlpTab { MlpP p[9]; };

// ---------------------------------------------------------------- utilities
__global__ void cvt_f16_kernel(const float* __restrict__ s, _Float16* __restrict__ d, int n) {
  int i = blockIdx.x * 256 + threadIdx.x;
  if (i < n) d[i] = (_Float16)s[i];
}

__global__ void h_init_kernel(const float* __restrict__ emb, const int* __restrict__ atom,
                              float* __restrict__ h) {
  int i = blockIdx.x * 256 + threadIdx.x;
  if (i >= NN * DD) return;
  int n = i >> 7, c = i & 127;
  h[i] = emb[atom[n] * DD + c];
}

// e_raw per edge + mean/var accumulation
__global__ void eraw_kernel(const float* __restrict__ pos, const int* __restrict__ ei,
                            float* __restrict__ e_raw, float* __restrict__ sums) {
  __shared__ float sAcc[2];
  if (threadIdx.x < 2) sAcc[threadIdx.x] = 0.f;
  __syncthreads();
  int e = blockIdx.x * 256 + threadIdx.x;
  float er = 0.f;
  if (e < EE) {
    int s = ei[e], d = ei[EE + e];
    float dx = pos[s*3+0]-pos[d*3+0], dy = pos[s*3+1]-pos[d*3+1], dz = pos[s*3+2]-pos[d*3+2];
    float el = sqrtf(dx*dx + dy*dy + dz*dz);
    er = (el - 2.7554f) * (1.f / 1.1664f);
    e_raw[e] = er;
  }
  float s1 = er, s2 = er * er;
  #pragma unroll
  for (int off = 16; off; off >>= 1) { s1 += __shfl_xor(s1, off, 32); s2 += __shfl_xor(s2, off, 32); }
  if ((threadIdx.x & 31) == 0) { atomicAdd(&sAcc[0], s1); atomicAdd(&sAcc[1], s2); }
  __syncthreads();
  if (threadIdx.x < 2) atomicAdd(&sums[threadIdx.x], sAcc[threadIdx.x]);
}

// BatchNorm of (W0*e + b0) over E edges folds to affine A*e + B (b0 cancels).
__global__ void bn_affine_kernel(const float* __restrict__ sums,
                                 const float* rW0, const float* rg0, const float* rbe0,
                                 const float* eW0, const float* eg0, const float* ebe0,
                                 float* __restrict__ affine) {
  float mean = sums[0] * (1.f / EE);
  float var  = sums[1] * (1.f / EE) - mean * mean;
  int j = threadIdx.x;
  if (j < 32) {
    float w = rW0[j];
    float A = w * rg0[j] * rsqrtf(w * w * var + EPSF);
    affine[j] = A; affine[32 + j] = rbe0[j] - mean * A;
  }
  if (j < 64) {
    float w = eW0[j];
    float A = w * eg0[j] * rsqrtf(w * w * var + EPSF);
    affine[64 + j] = A; affine[128 + j] = ebe0[j] - mean * A;
  }
}

// one wave32 per edge: router softmax, LB-loss stats, 9-way weighted scatter-add
__global__ __launch_bounds__(256) void edge_kernel(
    const float* __restrict__ e_raw, const int* __restrict__ ei, const float* __restrict__ h,
    const float* __restrict__ affine,
    const float* __restrict__ rW1, const float* __restrict__ rb1,
    const float* __restrict__ eW1, const float* __restrict__ eb1,
    float* __restrict__ acc, float* __restrict__ lb) {
  __shared__ float sLB[9];
  if (threadIdx.x < 9) sLB[threadIdx.x] = 0.f;
  __syncthreads();
  int wid = threadIdx.x >> 5, lane = threadIdx.x & 31;
  int edge = blockIdx.x * 8 + wid;
  float er = e_raw[edge];
  // router logit for k = lane&7 (each group of 8 lanes computes the full softmax)
  int k = lane & 7;
  float logit = rb1[k];
  {
    const float* w = rW1 + k * 32;
    #pragma unroll 8
    for (int j = 0; j < 32; ++j) {
      float hv = fmaf(affine[j], er, affine[32 + j]);
      hv = hv >= 0.f ? hv : 0.01f * hv;
      logit = fmaf(w[j], hv, logit);
    }
  }
  float mx = logit;
  #pragma unroll
  for (int off = 4; off; off >>= 1) mx = fmaxf(mx, __shfl_xor(mx, off, 8));
  float ex = __expf(logit - mx);
  float sm = ex;
  #pragma unroll
  for (int off = 4; off; off >>= 1) sm += __shfl_xor(sm, off, 8);
  float route = ex / sm;                      // route[k] on lane with lane&7==k
  if (lane < 8) atomicAdd(&sLB[lane], route); // per-k mean accumulation
  float sq = route * route;
  #pragma unroll
  for (int off = 4; off; off >>= 1) sq += __shfl_xor(sq, off, 8);
  if (lane == 0) atomicAdd(&sLB[8], sq);
  // edge feature MLP: lane l computes output channel l (DE=32)
  float ef = eb1[lane];
  {
    const float* w = eW1 + lane * 64;
    #pragma unroll 8
    for (int j = 0; j < 64; ++j) {
      float hv = fmaf(affine[64 + j], er, affine[128 + j]);
      hv = hv >= 0.f ? hv : 0.01f * hv;
      ef = fmaf(w[j], hv, ef);
    }
  }
  int src = ei[edge], dst = ei[EE + edge];
  const float4* hs = (const float4*)(h + (size_t)src * DD);
  float4 xs = hs[lane];
  size_t dbase = (size_t)dst * 160;
  #pragma unroll
  for (int widx = 0; widx < 9; ++widx) {
    float w = (widx == 0) ? 1.f : __shfl(route, widx - 1, 8);
    float* a = acc + (size_t)widx * (size_t)NN * 160 + dbase;
    atomicAdd(a + lane * 4 + 0, xs.x * w);
    atomicAdd(a + lane * 4 + 1, xs.y * w);
    atomicAdd(a + lane * 4 + 2, xs.z * w);
    atomicAdd(a + lane * 4 + 3, xs.w * w);
    atomicAdd(a + 128 + lane, ef * w);
  }
  __syncthreads();
  if (threadIdx.x < 9) atomicAdd(&lb[threadIdx.x], sLB[threadIdx.x]);
}

// WMMA node MLP, 64-row blocks: X[64,160] @ W0^T -> LN -> act -> @ W1^T -> out.
// X tile staged once in LDS as f16 (168-half row stride: conflict-free fragment
// gathers); B fragments live in registers, each reused by 4 row-tile WMMAs.
__global__ __launch_bounds__(256) void node_mlp_kernel(const float* __restrict__ acc,
                                                       MlpTab tab, float* __restrict__ out) {
  const int widx = blockIdx.y;
  const int row0 = blockIdx.x * 64;
  const MlpP P = tab.p[widx];
  __shared__ _Float16 Fh[64][168];   // X tile (f16); reused for post-LN activations
  __shared__ float    Ps[64][132];   // pre-activation (f32)
  const float* xsrc = acc + ((size_t)widx * (size_t)NN + row0) * 160;
  for (int u = threadIdx.x; u < 64 * 80; u += 256) {
    int r = u / 80, cc = (u % 80) * 2;
    float2 xv = *(const float2*)(xsrc + r * 160 + cc);
    Fh[r][cc]     = (_Float16)xv.x;
    Fh[r][cc + 1] = (_Float16)xv.y;
  }
  int wv = threadIdx.x >> 5, lane = threadIdx.x & 31;
  int m = lane & 15, hi = lane >> 4;
  int ncol = wv * 16 + m;
  const unsigned* w0u = (const unsigned*)(P.W0 + (size_t)ncol * 160);
  const unsigned* w1u = (const unsigned*)(P.W1 + (size_t)ncol * 128);
  __builtin_prefetch(w1u, 0, 1);
  float gq[4], beq[4];
  #pragma unroll
  for (int q = 0; q < 4; ++q) { gq[q] = P.g0[lane + 32 * q]; beq[q] = P.be0[lane + 32 * q]; }
  __syncthreads();
  // ---- GEMM1: K = 160 = 5 x 32, 4 row-tiles share each B fragment
  v8f c[4];
  #pragma unroll
  for (int t = 0; t < 4; ++t)
    #pragma unroll
    for (int r = 0; r < 8; ++r) c[t][r] = 0.f;
  #pragma unroll
  for (int kk = 0; kk < 5; ++kk) {
    int kb = kk * 32;
    HU b;
    #pragma unroll
    for (int v = 0; v < 8; ++v) b.u[v] = w0u[(kb >> 1) + hi * 8 + v];   // B: k = kb + 16*hi + s (contiguous, same order as A)
    #pragma unroll
    for (int t = 0; t < 4; ++t) {
      HU a;
      #pragma unroll
      for (int v = 0; v < 8; ++v) {
        int Ka = kb + hi * 16 + v * 2;        // A-layout
        a.u[v] = *(const unsigned*)&Fh[t * 16 + m][Ka];
      }
      c[t] = __builtin_amdgcn_wmma_f32_16x16x32_f16(false, a.h, false, b.h, (short)0, c[t], false, false);
    }
  }
  float bias0 = P.b0[ncol];
  #pragma unroll
  for (int t = 0; t < 4; ++t)
    #pragma unroll
    for (int r = 0; r < 8; ++r) Ps[t * 16 + r + hi * 8][ncol] = c[t][r] + bias0;
  __syncthreads();
  // ---- LayerNorm + act: wave wv handles rows 8wv..8wv+7; write f16 acts to Fh
  const bool isShared = (widx == 0);
  for (int rr = 0; rr < 8; ++rr) {
    int row = wv * 8 + rr;
    float s = 0.f, s2 = 0.f;
    #pragma unroll
    for (int q = 0; q < 4; ++q) { float v = Ps[row][lane + 32 * q]; s += v; s2 += v * v; }
    #pragma unroll
    for (int off = 16; off; off >>= 1) { s += __shfl_xor(s, off, 32); s2 += __shfl_xor(s2, off, 32); }
    float mean = s * (1.f / 128.f);
    float var  = s2 * (1.f / 128.f) - mean * mean;
    float inv  = rsqrtf(var + EPSF);
    #pragma unroll
    for (int q = 0; q < 4; ++q) {
      int j = lane + 32 * q;
      float v = (Ps[row][j] - mean) * inv * gq[q] + beq[q];
      v = (v >= 0.f) ? v : (isShared ? 0.01f * v : 0.f);
      Fh[row][j] = (_Float16)v;
    }
  }
  __syncthreads();
  // ---- GEMM2: K = 128 = 4 x 32
  v8f d[4];
  #pragma unroll
  for (int t = 0; t < 4; ++t)
    #pragma unroll
    for (int r = 0; r < 8; ++r) d[t][r] = 0.f;
  #pragma unroll
  for (int kk = 0; kk < 4; ++kk) {
    int kb = kk * 32;
    HU b;
    #pragma unroll
    for (int v = 0; v < 8; ++v) b.u[v] = w1u[(kb >> 1) + hi * 8 + v];
    #pragma unroll
    for (int t = 0; t < 4; ++t) {
      HU a;
      #pragma unroll
      for (int v = 0; v < 8; ++v) {
        int Ka = kb + hi * 16 + v * 2;
        a.u[v] = *(const unsigned*)&Fh[t * 16 + m][Ka];
      }
      d[t] = __builtin_amdgcn_wmma_f32_16x16x32_f16(false, a.h, false, b.h, (short)0, d[t], false, false);
    }
  }
  float bias1 = P.b1[ncol];
  #pragma unroll
  for (int t = 0; t < 4; ++t)
    #pragma unroll
    for (int r = 0; r < 8; ++r)
      atomicAdd(&out[(size_t)(row0 + t * 16 + r + hi * 8) * DD + ncol], d[t][r] + bias1);
}

// -------- graph norm / residual / readout
__global__ void gn_sum_kernel(const float* __restrict__ x, const int* __restrict__ batch,
                              float* __restrict__ gsum, float* __restrict__ gcnt) {
  int i = blockIdx.x * 256 + threadIdx.x;
  if (i >= NN * DD) return;
  int n = i >> 7, c = i & 127, b = batch[n];
  atomicAdd(&gsum[b * DD + c], x[i]);
  if (c == 0) atomicAdd(&gcnt[b], 1.f);
}

__global__ void gn_center_kernel(float* __restrict__ out, const int* __restrict__ batch,
                                 const float* __restrict__ gsum, const float* __restrict__ gcnt,
                                 const float* __restrict__ ms, float* __restrict__ gsq) {
  int i = blockIdx.x * 256 + threadIdx.x;
  if (i >= NN * DD) return;
  int n = i >> 7, c = i & 127, b = batch[n];
  float mean = gsum[b * DD + c] / gcnt[b];
  float oc = out[i] - mean * ms[c];
  out[i] = oc;
  atomicAdd(&gsq[b * DD + c], oc * oc);
}

__global__ void gn_apply_kernel(const float* __restrict__ out, const int* __restrict__ batch,
                                const float* __restrict__ gsq, const float* __restrict__ gcnt,
                                const float* __restrict__ g, const float* __restrict__ be,
                                const float* __restrict__ alpha, int aidx, float* __restrict__ h) {
  int i = blockIdx.x * 256 + threadIdx.x;
  if (i >= NN * DD) return;
  int n = i >> 7, c = i & 127, b = batch[n];
  float var = gsq[b * DD + c] / gcnt[b];
  float v = out[i] * rsqrtf(var + EPSF) * g[c] + be[c];
  v = fmaxf(v, 0.f);
  h[i] += alpha[aidx] * v;
}

__global__ void h_add_kernel(const float* __restrict__ out, const float* __restrict__ alpha,
                             int aidx, float* __restrict__ h) {
  int i = blockIdx.x * 256 + threadIdx.x;
  if (i < NN * DD) h[i] += alpha[aidx] * out[i];
}

__global__ void final_kernel(const float* __restrict__ gsum, const float* __restrict__ gcnt,
                             const float* __restrict__ lb, float* __restrict__ outp) {
  int g = threadIdx.x;
  if (g < GG) {
    float s = 0.f;
    for (int c = 0; c < DD; ++c) s += gsum[g * DD + c];
    outp[g] = s / (gcnt[g] * (float)DD);
  }
  if (g == 0) {
    float total = 0.f;
    for (int i = 0; i < 2; ++i) {
      const float* L = lb + i * 9;
      float s2 = 0.f;
      for (int k = 0; k < KK; ++k) { float mm = L[k] * (1.f / EE); s2 += mm * mm; }
      const float frac = 1.f / KK;
      float ebl = (s2 - frac) * (1.f / (1.f - frac));
      float ul  = 1.f - L[8] * (1.f / EE);
      float unf = 0.1f * ebl + 0.1f * ul;
      const float t = 0.2f * 0.1f;
      float lbl = (fmaxf(unf, t) - t) * (0.2f / (0.2f - t));
      total += 0.5f * lbl;
    }
    outp[GG] = total;
  }
}

// ---------------------------------------------------------------- host
extern "C" void kernel_launch(void* const* d_in, const int* in_sizes, int n_in,
                              void* d_out, int out_size, void* d_ws, size_t ws_size,
                              hipStream_t stream) {
  (void)in_sizes; (void)n_in; (void)out_size; (void)ws_size;
  const int*   atom  = (const int*)d_in[0];
  const float* pos   = (const float*)d_in[1];
  const int*   ei    = (const int*)d_in[2];
  const int*   batch = (const int*)d_in[3];
  auto F = [&](int i) { return (const float*)d_in[i]; };
  const float* alpha = F(4);
  // params flatten (sorted dict keys): alpha, convs[{edge,experts[8],router,shared} x {W0,W1,b0,b1,be0,g0}], emb, gn
  // conv0 base 5, conv1 base 71; emb 137; gn[0]=138..140, gn[1]=141..143
  float* ws      = (float*)d_ws;
  float* sums    = ws;                 // 2
  float* lb      = ws + 2;             // 2 x 9
  float* affine  = ws + 20;            // 192
  float* gcnt    = ws + 212;           // 50
  float* gsum    = ws + 262;           // 6400
  float* gsq     = ws + 6662;          // 6400
  float* e_raw   = ws + 13072;         // EE
  float* h       = ws + 653072;        // NN*128
  float* outb    = ws + 5773072;       // NN*128
  _Float16* hbuf = (_Float16*)(ws + 10893072);   // 18 x (20480 + 16384) halves
  float* acc     = ws + 11224848;      // 9 * NN * 160

  hipMemsetAsync(ws, 0, 20 * sizeof(float), stream);
  h_init_kernel<<<(NN * DD + 255) / 256, 256, 0, stream>>>(F(137), atom, h);
  eraw_kernel<<<EE / 256, 256, 0, stream>>>(pos, ei, e_raw, sums);

  // convert node-MLP weights to f16 (mlp 0 = shared, 1..8 = experts)
  for (int cv = 0; cv < 2; ++cv) {
    int cb = cv ? 71 : 5;
    for (int mI = 0; mI < 9; ++mI) {
      int base = (mI == 0) ? (cb + 60) : (cb + 6 + 6 * (mI - 1));
      _Float16* w0h = hbuf + (size_t)(cv * 9 + mI) * 36864;
      _Float16* w1h = w0h + 20480;
      cvt_f16_kernel<<<(20480 + 255) / 256, 256, 0, stream>>>(F(base), w0h, 20480);
      cvt_f16_kernel<<<(16384 + 255) / 256, 256, 0, stream>>>(F(base + 1), w1h, 16384);
    }
  }

  for (int cv = 0; cv < 2; ++cv) {
    int cb = cv ? 71 : 5;
    // router base cb+54, edge base cb
    bn_affine_kernel<<<1, 64, 0, stream>>>(sums, F(cb + 54), F(cb + 59), F(cb + 58),
                                           F(cb), F(cb + 5), F(cb + 4), affine);
    hipMemsetAsync(acc, 0, (size_t)9 * NN * 160 * sizeof(float), stream);
    edge_kernel<<<EE / 8, 256, 0, stream>>>(e_raw, ei, h, affine,
                                            F(cb + 55), F(cb + 57), F(cb + 1), F(cb + 3),
                                            acc, lb + cv * 9);
    hipMemsetAsync(outb, 0, (size_t)NN * DD * sizeof(float), stream);
    MlpTab tab;
    for (int mI = 0; mI < 9; ++mI) {
      int base = (mI == 0) ? (cb + 60) : (cb + 6 + 6 * (mI - 1));
      _Float16* w0h = hbuf + (size_t)(cv * 9 + mI) * 36864;
      tab.p[mI] = { w0h, w0h + 20480, F(base + 2), F(base + 3), F(base + 5), F(base + 4) };
    }
    node_mlp_kernel<<<dim3(NN / 64, 9), 256, 0, stream>>>(acc, tab, outb);
    int nb = (NN * DD + 255) / 256;
    if (cv == 0) {
      hipMemsetAsync(gcnt, 0, (50 + 6400 + 6400) * sizeof(float), stream);
      gn_sum_kernel<<<nb, 256, 0, stream>>>(outb, batch, gsum, gcnt);
      gn_center_kernel<<<nb, 256, 0, stream>>>(outb, batch, gsum, gcnt, F(140), gsq);
      gn_apply_kernel<<<nb, 256, 0, stream>>>(outb, batch, gsq, gcnt, F(138), F(139), alpha, 0, h);
    } else {
      h_add_kernel<<<nb, 256, 0, stream>>>(outb, alpha, 1, h);
    }
  }

  hipMemsetAsync(gcnt, 0, (50 + 6400) * sizeof(float), stream);
  gn_sum_kernel<<<(NN * DD + 255) / 256, 256, 0, stream>>>(h, batch, gsum, gcnt);
  final_kernel<<<1, 64, 0, stream>>>(gsum, gcnt, lb, (float*)d_out);
}
